// GATConv_32332513804854
// MI455X (gfx1250) — matrix operand order repaired and checked by reading, not compiled, not run on hardware
//
#include <hip/hip_runtime.h>
#include <math.h>

typedef __attribute__((ext_vector_type(16))) _Float16 v16h;
typedef __attribute__((ext_vector_type(8)))  float    v8f;

#define INCH  128
#define HEADS 4
#define CPH   32
#define HC    128
#define NEG_SLOPE 0.2f

// ---------------- K0: Bezier parameter interpolation ----------------
__global__ void k_params(const float* __restrict__ coeffs,
                         const float* __restrict__ lw,   // [3,128,128]
                         const float* __restrict__ lb,   // [3,128]
                         const float* __restrict__ av_s, // [3,128]
                         const float* __restrict__ av_d, // [3,128]
                         _Float16* __restrict__ Wh,      // [128,128] f16
                         float* __restrict__ bias,
                         float* __restrict__ a_s,
                         float* __restrict__ a_d) {
  int i = blockIdx.x * blockDim.x + threadIdx.x;
  float c0 = coeffs[0], c1 = coeffs[1], c2 = coeffs[2];
  if (i < HC * INCH) {
    float w = c0 * lw[i] + c1 * lw[HC * INCH + i] + c2 * lw[2 * HC * INCH + i];
    Wh[i] = (_Float16)w;
  }
  if (i < HC) {
    bias[i] = c0 * lb[i]   + c1 * lb[HC + i]   + c2 * lb[2 * HC + i];
    a_s[i]  = c0 * av_s[i] + c1 * av_s[HC + i] + c2 * av_s[2 * HC + i];
    a_d[i]  = c0 * av_d[i] + c1 * av_d[HC + i] + c2 * av_d[2 * HC + i];
  }
}

// ---------------- K1: init accumulators ----------------
__global__ void k_init(float* __restrict__ out, float* __restrict__ amax,
                       float* __restrict__ denom, int n) {
  int i = blockIdx.x * blockDim.x + threadIdx.x;
  if (i < n * HC) out[i] = 0.0f;
  if (i < n * HEADS) { amax[i] = -INFINITY; denom[i] = 0.0f; }
}

// ---------------- K2: WMMA GEMM  xt[N,128] = x[N,128] @ W^T + bias ----------------
// Weights staged once per block into LDS (32 KB); one wave computes a 16-row
// tile across all 128 output channels: 8 N-tiles x 4 K-steps = 32 WMMAs.
__global__ void __launch_bounds__(256)
k_gemm(const float* __restrict__ x, const _Float16* __restrict__ Wh,
       const float* __restrict__ bias, float* __restrict__ xt, int n) {
  __shared__ _Float16 sW[HC * INCH];           // 32 KB

  // Stage Wh -> LDS (all 256 threads, 8 x 16B each)
  {
    const float4* src = (const float4*)Wh;     // 2048 x float4
    float4* dst = (float4*)sW;
#pragma unroll
    for (int i = 0; i < 8; i++) dst[threadIdx.x + i * 256] = src[threadIdx.x + i * 256];
  }
  __syncthreads();

  const int lane  = threadIdx.x & 31;
  const int wave  = threadIdx.x >> 5;
  const int tile  = blockIdx.x * 8 + wave;
  const int mbase = tile * 16;
  if (mbase < n) {                             // wave-uniform guard AFTER barrier
    const int  lm   = lane & 15;
    const bool hi   = lane >= 16;
    const int  kb8  = hi ? 8 : 0;              // table-order half-wave K base; unused (A and B use kb16)
    const int  kb16 = hi ? 16 : 0;             // A and B half-wave K base, contiguous k = 16h + s

    v8f acc[8] = {};

    const int m = (mbase + lm < n) ? (mbase + lm) : (n - 1);
    const float* xrow = x + (size_t)m * INCH;

#pragma unroll
    for (int ks = 0; ks < 4; ks++) {
      const int k0 = ks * 32;
      const float* p = xrow + k0 + kb16;
      float4 f0 = *(const float4*)(p);
      float4 f1 = *(const float4*)(p + 4);
      float4 f2 = *(const float4*)(p + 8);
      float4 f3 = *(const float4*)(p + 12);
      v16h a;
      a[0]  = (_Float16)f0.x; a[1]  = (_Float16)f0.y; a[2]  = (_Float16)f0.z; a[3]  = (_Float16)f0.w;
      a[4]  = (_Float16)f1.x; a[5]  = (_Float16)f1.y; a[6]  = (_Float16)f1.z; a[7]  = (_Float16)f1.w;
      a[8]  = (_Float16)f2.x; a[9]  = (_Float16)f2.y; a[10] = (_Float16)f2.z; a[11] = (_Float16)f2.w;
      a[12] = (_Float16)f3.x; a[13] = (_Float16)f3.y; a[14] = (_Float16)f3.z; a[15] = (_Float16)f3.w;
#pragma unroll
      for (int t = 0; t < 8; t++) {
        const int nn = t * 16 + lm;            // output channel for this lane
        v16h b = *(const v16h*)(&sW[nn * INCH + k0 + kb16]);   // ds_load
        acc[t] = __builtin_amdgcn_wmma_f32_16x16x32_f16(
            false, a, false, b, (short)0, acc[t], false, false);
      }
    }

    // Epilogue: D layout -> lane gives N, VGPR r gives M (upper lanes +8); add bias.
    const int mrow = mbase + (hi ? 8 : 0);
#pragma unroll
    for (int t = 0; t < 8; t++) {
      const int nn = t * 16 + lm;
      const float bv = bias[nn];
#pragma unroll
      for (int r = 0; r < 8; r++) {
        const int mm = mrow + r;
        if (mm < n) xt[(size_t)mm * HC + nn] = acc[t][r] + bv;
      }
    }
  }
}

// ---------------- K3: per-node attention dot products ----------------
__global__ void k_alpha(const float* __restrict__ xt,
                        const float* __restrict__ a_s, const float* __restrict__ a_d,
                        float* __restrict__ as_out, float* __restrict__ ad_out, int n) {
  int t = blockIdx.x * blockDim.x + threadIdx.x;
  if (t >= n * HEADS) return;
  int node = t >> 2, h = t & 3;
  const float4* xr = (const float4*)(xt + (size_t)node * HC + h * CPH);
  const float4* u  = (const float4*)(a_s + h * CPH);
  const float4* v  = (const float4*)(a_d + h * CPH);
  float s1 = 0.0f, s2 = 0.0f;
#pragma unroll
  for (int i = 0; i < 8; i++) {
    float4 a = xr[i], b = u[i], c = v[i];
    s1 += a.x * b.x + a.y * b.y + a.z * b.z + a.w * b.w;
    s2 += a.x * c.x + a.y * c.y + a.z * c.z + a.w * c.w;
  }
  as_out[t] = s1;
  ad_out[t] = s2;
}

// Float atomic max via sign-aware int/uint atomics (monotone bit patterns).
__device__ inline void atomicMaxFloat(float* addr, float val) {
  if (val >= 0.0f) atomicMax((int*)addr, __float_as_int(val));
  else             atomicMin((unsigned int*)addr, __float_as_uint(val));
}

// ---------------- K4: edge leaky-relu + segment max ----------------
__global__ void k_edge1(const int* __restrict__ row, const int* __restrict__ col,
                        const float* __restrict__ as, const float* __restrict__ ad,
                        float* __restrict__ alphaE, float* __restrict__ amax, int e) {
  int t = blockIdx.x * blockDim.x + threadIdx.x;
  if (t >= e * HEADS) return;
  int ed = t >> 2, h = t & 3;
  int r = row[ed], c = col[ed];
  float a = as[r * HEADS + h] + ad[c * HEADS + h];
  a = a > 0.0f ? a : NEG_SLOPE * a;
  alphaE[t] = a;
  atomicMaxFloat(&amax[c * HEADS + h], a);
}

// ---------------- K5: exp + segment sum ----------------
__global__ void k_edge2(const int* __restrict__ col, float* __restrict__ alphaE,
                        const float* __restrict__ amax, float* __restrict__ denom, int e) {
  int t = blockIdx.x * blockDim.x + threadIdx.x;
  if (t >= e * HEADS) return;
  int ed = t >> 2, h = t & 3;
  int c = col[ed];
  float ex = __expf(alphaE[t] - amax[c * HEADS + h]);
  alphaE[t] = ex;
  atomicAdd(&denom[c * HEADS + h], ex);
}

// ---------------- K6: normalize + weighted scatter-add ----------------
// One wave = one (edge, head). Wave index forced into SGPRs via readfirstlane
// so row/col/alpha/denom fetches become scalar loads; the per-lane work is a
// coalesced 128B xt read + coalesced 128B-line global_atomic_add_f32.
__global__ void __launch_bounds__(256)
k_edge3(const int* __restrict__ row, const int* __restrict__ col,
        const float* __restrict__ alphaE, const float* __restrict__ denom,
        const float* __restrict__ xt, float* __restrict__ out, int e) {
  const int lane = threadIdx.x & 31;
  const int wvu = __builtin_amdgcn_readfirstlane((blockIdx.x << 3) | (threadIdx.x >> 5));
  if (wvu >= e * HEADS) return;                // uniform guard
  const int ed = wvu >> 2;
  const int h  = wvu & 3;
  const int r = row[ed];                       // scalar loads (SGPR address)
  const int c = col[ed];
  const float w = alphaE[ed * HEADS + h] / (denom[c * HEADS + h] + 1e-16f);
  const int ch = (h << 5) | lane;
  atomicAdd(&out[(size_t)c * HC + ch], xt[(size_t)r * HC + ch] * w);
}

extern "C" void kernel_launch(void* const* d_in, const int* in_sizes, int n_in,
                              void* d_out, int out_size, void* d_ws, size_t ws_size,
                              hipStream_t stream) {
  const float* x      = (const float*)d_in[0];
  const int*   ei     = (const int*)d_in[1];
  const float* coeffs = (const float*)d_in[2];
  const float* lw     = (const float*)d_in[3];
  const float* lb     = (const float*)d_in[4];
  const float* av_s   = (const float*)d_in[5];
  const float* av_d   = (const float*)d_in[6];
  float* out = (float*)d_out;

  const int n = in_sizes[0] / INCH;  // 50000
  const int e = in_sizes[1] / 2;     // 600000
  const int* row = ei;
  const int* col = ei + e;

  char* ws = (char*)d_ws;
  size_t off = 0;
  auto alloc = [&](size_t bytes) -> void* {
    void* p = ws + off;
    off = (off + bytes + 255) & ~(size_t)255;
    return p;
  };
  _Float16* Wh   = (_Float16*)alloc((size_t)HC * INCH * sizeof(_Float16));
  float* bias    = (float*)alloc(HC * sizeof(float));
  float* a_s     = (float*)alloc(HC * sizeof(float));
  float* a_d     = (float*)alloc(HC * sizeof(float));
  float* xt      = (float*)alloc((size_t)n * HC * sizeof(float));
  float* asrc    = (float*)alloc((size_t)n * HEADS * sizeof(float));
  float* adst    = (float*)alloc((size_t)n * HEADS * sizeof(float));
  float* amax    = (float*)alloc((size_t)n * HEADS * sizeof(float));
  float* denom   = (float*)alloc((size_t)n * HEADS * sizeof(float));
  float* alphaE  = (float*)alloc((size_t)e * HEADS * sizeof(float));

  k_params<<<(HC * INCH + 255) / 256, 256, 0, stream>>>(coeffs, lw, lb, av_s, av_d,
                                                        Wh, bias, a_s, a_d);
  k_init<<<((unsigned)(((size_t)n * HC + 255) / 256)), 256, 0, stream>>>(out, amax, denom, n);

  const int tiles = (n + 15) / 16;
  k_gemm<<<(tiles + 7) / 8, 256, 0, stream>>>(x, Wh, bias, xt, n);

  k_alpha<<<(n * HEADS + 255) / 256, 256, 0, stream>>>(xt, a_s, a_d, asrc, adst, n);
  k_edge1<<<(e * HEADS + 255) / 256, 256, 0, stream>>>(row, col, asrc, adst, alphaE, amax, e);
  k_edge2<<<(e * HEADS + 255) / 256, 256, 0, stream>>>(col, alphaE, amax, denom, e);

  const int waves6 = e * HEADS;                      // one wave per (edge, head)
  k_edge3<<<(waves6 + 7) / 8, 256, 0, stream>>>(row, col, alphaE, denom, xt, out, e);
}
